// MafExtractor_44667659879175
// MI455X (gfx1250) — matrix operand order repaired and checked by reading, not compiled, not run on hardware
//
#include <hip/hip_runtime.h>
#include <hip/hip_bf16.h>
#include <math.h>

typedef __attribute__((ext_vector_type(16))) _Float16 v16h;
typedef __attribute__((ext_vector_type(8)))  float    v8f;

#define BB   32
#define LL   16384
#define HH   128

#define LW0  16375
#define LW1  16371
#define LW2  16367
#define OFF0 5
#define OFF1 7
#define OFF2 9

__device__ __forceinline__ float gelu_exact(float x) {
    return 0.5f * x * (1.0f + erff(x * 0.7071067811865475f));
}

// ---------------------------------------------------------------------------
// Kernel 1: 3 valid 1D convs (w=10,14,18 over AA=8 channels) + cls/reg heads
// One thread per (b,pos); windows overlap so x reads hit WGP$/L2.
// ---------------------------------------------------------------------------
__global__ __launch_bounds__(256) void conv_heads_kernel(
    const float* __restrict__ x,
    const float* __restrict__ cw0, const float* __restrict__ cw1, const float* __restrict__ cw2,
    const float* __restrict__ conv_b,
    const float* __restrict__ cls_w, const float* __restrict__ cls_b,
    const float* __restrict__ reg_w, const float* __restrict__ reg_b,
    float* __restrict__ cls0, float* __restrict__ cls1, float* __restrict__ cls2,
    float* __restrict__ reg0, float* __restrict__ reg1, float* __restrict__ reg2,
    float* __restrict__ mask)
{
    __shared__ float sw[336];     // 80 + 112 + 144 conv weights
    __shared__ float shead[15];   // conv_b, cls_w, cls_b, reg_w, reg_b (3 each)
    for (int i = threadIdx.x; i < 336; i += 256)
        sw[i] = (i < 80) ? cw0[i] : (i < 192) ? cw1[i - 80] : cw2[i - 192];
    if (threadIdx.x < 15) {
        int j = threadIdx.x % 3, g = threadIdx.x / 3;
        const float* p = (g == 0) ? conv_b : (g == 1) ? cls_w : (g == 2) ? cls_b
                       : (g == 3) ? reg_w : reg_b;
        shead[threadIdx.x] = p[j];
    }
    __syncthreads();

    int idx = blockIdx.x * 256 + threadIdx.x;       // exactly B*L threads
    int b = idx >> 14, pos = idx & (LL - 1);
    const float* xb = x + ((long long)b * LL + pos) * 8;

    float a0 = 0.f, a1 = 0.f, a2 = 0.f;
    #pragma unroll
    for (int k = 0; k < 18; ++k) {
        if (pos + k >= LL) break;
        float4 u = *(const float4*)(xb + k * 8);
        float4 v = *(const float4*)(xb + k * 8 + 4);
        float xv[8] = {u.x, u.y, u.z, u.w, v.x, v.y, v.z, v.w};
        #pragma unroll
        for (int c = 0; c < 8; ++c) {
            float xc = xv[c];
            if (k < 10) a0 += xc * sw[c * 10 + k];
            if (k < 14) a1 += xc * sw[80 + c * 14 + k];
            a2 += xc * sw[192 + c * 18 + k];
        }
    }
    float y0 = a0 + shead[0], y1 = a1 + shead[1], y2 = a2 + shead[2];
    if (pos < LW0) { long long o = (long long)b * LW0 + pos;
        cls0[o] = y0 * shead[3] + shead[6];  reg0[o] = y0 * shead[9]  + shead[12]; }
    if (pos < LW1) { long long o = (long long)b * LW1 + pos;
        cls1[o] = y1 * shead[4] + shead[7];  reg1[o] = y1 * shead[10] + shead[13]; }
    if (pos < LW2) { long long o = (long long)b * LW2 + pos;
        cls2[o] = y2 * shead[5] + shead[8];  reg2[o] = y2 * shead[11] + shead[14]; }
    mask[idx] = 1.0f;  // lengths == L path: all-true mask
}

// ---------------------------------------------------------------------------
// Kernel 2: per-(b,scale) pooling: min / mean / top-3 mean of cls scores.
// One block per (b, j); block reduction in LDS.
// ---------------------------------------------------------------------------
__global__ __launch_bounds__(256) void pool_kernel(
    const float* __restrict__ cls0, const float* __restrict__ cls1, const float* __restrict__ cls2,
    float* __restrict__ gfeat)
{
    int b = blockIdx.x / 3, j = blockIdx.x % 3;
    const float* src; int Lw;
    if (j == 0)      { src = cls0 + (long long)b * LW0; Lw = LW0; }
    else if (j == 1) { src = cls1 + (long long)b * LW1; Lw = LW1; }
    else             { src = cls2 + (long long)b * LW2; Lw = LW2; }

    float mn = 3.4e38f, sum = 0.f;
    float t0 = -3.4e38f, t1 = -3.4e38f, t2 = -3.4e38f;
    for (int i = threadIdx.x; i < Lw; i += 256) {
        float v = src[i];
        mn = fminf(mn, v);
        sum += v;
        if (v > t0)      { t2 = t1; t1 = t0; t0 = v; }
        else if (v > t1) { t2 = t1; t1 = v; }
        else if (v > t2) { t2 = v; }
    }
    __shared__ float smin[256], ssum[256], st[256][3];
    smin[threadIdx.x] = mn; ssum[threadIdx.x] = sum;
    st[threadIdx.x][0] = t0; st[threadIdx.x][1] = t1; st[threadIdx.x][2] = t2;
    __syncthreads();
    for (int s = 128; s > 0; s >>= 1) {
        if (threadIdx.x < s) {
            smin[threadIdx.x] = fminf(smin[threadIdx.x], smin[threadIdx.x + s]);
            ssum[threadIdx.x] += ssum[threadIdx.x + s];
            float a0 = st[threadIdx.x][0], a1 = st[threadIdx.x][1], a2 = st[threadIdx.x][2];
            #pragma unroll
            for (int q = 0; q < 3; ++q) {
                float v = st[threadIdx.x + s][q];
                if (v > a0)      { a2 = a1; a1 = a0; a0 = v; }
                else if (v > a1) { a2 = a1; a1 = v; }
                else if (v > a2) { a2 = v; }
            }
            st[threadIdx.x][0] = a0; st[threadIdx.x][1] = a1; st[threadIdx.x][2] = a2;
        }
        __syncthreads();
    }
    if (threadIdx.x == 0) {
        gfeat[b * 9 + j * 3 + 0] = smin[0];
        gfeat[b * 9 + j * 3 + 1] = ssum[0] / (float)Lw;
        gfeat[b * 9 + j * 3 + 2] = (st[0][0] + st[0][1] + st[0][2]) * (1.f / 3.f);
    }
}

// ---------------------------------------------------------------------------
// Kernel 3: global MLP  [32,9] -> GELU(128) -> [32,128]. Single block, LDS.
// ---------------------------------------------------------------------------
__global__ __launch_bounds__(256) void gmlp_kernel(
    const float* __restrict__ gfeat,
    const float* __restrict__ pw1, const float* __restrict__ pb1,
    const float* __restrict__ pw2, const float* __restrict__ pb2,
    float* __restrict__ gout)
{
    __shared__ float sg[BB * 9];
    __shared__ float sw1[9 * HH];
    __shared__ float sb1[HH];
    __shared__ float shid[BB * HH];
    for (int i = threadIdx.x; i < BB * 9; i += 256) sg[i] = gfeat[i];
    for (int i = threadIdx.x; i < 9 * HH; i += 256) sw1[i] = pw1[i];
    for (int i = threadIdx.x; i < HH;     i += 256) sb1[i] = pb1[i];
    __syncthreads();
    for (int i = threadIdx.x; i < BB * HH; i += 256) {
        int b = i >> 7, n = i & 127;
        float s = sb1[n];
        #pragma unroll
        for (int t = 0; t < 9; ++t) s += sg[b * 9 + t] * sw1[t * HH + n];
        shid[i] = gelu_exact(s);
    }
    __syncthreads();
    for (int i = threadIdx.x; i < BB * HH; i += 256) {
        int b = i >> 7, n = i & 127;
        float s = pb2[n];
        for (int k = 0; k < HH; ++k) s += shid[b * HH + k] * pw2[k * HH + n];
        gout[i] = s;
    }
}

// ---------------------------------------------------------------------------
// Kernel 4: local MLP with WMMA.
//   feats[m,0:3] (gathered from cls scores with center-aligned offsets)
//   h = GELU(feats @ lp_w1 + lp_b1)          -- computed in registers, f16
//   out = h @ lp_w2 + lp_b2                  -- v_wmma_f32_16x16x32_f16
// lp_w2 pre-swizzled into B-fragment order in LDS (32 KB): each fragment
// read is 32 contiguous bytes per lane (2x ds_load_b128).
// Wave = 16-row M tile; 8 N-tiles x 4 K-steps = 32 WMMAs per wave.
// ---------------------------------------------------------------------------
__global__ __launch_bounds__(256) void local_mlp_wmma_kernel(
    const float* __restrict__ cls0, const float* __restrict__ cls1, const float* __restrict__ cls2,
    const float* __restrict__ lw1, const float* __restrict__ lb1,
    const float* __restrict__ lw2, const float* __restrict__ lb2,
    float* __restrict__ out)
{
    __shared__ __align__(32) _Float16 w2f[4 * 8 * 32 * 16];  // [kk][nt][lane][e]
    __shared__ float sw1[3 * HH];
    __shared__ float sb1[HH];
    __shared__ float sb2[HH];

    // Pre-swizzle lp_w2[k][n] (f32) into WMMA B-fragment layout (f16):
// B 32x16 f16 layout: lane lo = column n, K = (e<8 ? e : e+8) + hi*8.
    for (int f = threadIdx.x; f < 16384; f += 256) {
        int e    = f & 15;
        int ln   = (f >> 4) & 31;
        int nt   = (f >> 9) & 7;
        int kk   = f >> 12;
        int k    = kk * 32 + (e < 8 ? e : e + 8) + ((ln >> 4) << 3);
        int n    = nt * 16 + (ln & 15);
        w2f[f] = (_Float16)lw2[k * HH + n];
    }
    for (int i = threadIdx.x; i < 3 * HH; i += 256) sw1[i] = lw1[i];
    for (int i = threadIdx.x; i < HH;     i += 256) { sb1[i] = lb1[i]; sb2[i] = lb2[i]; }
    __syncthreads();

    int wave = threadIdx.x >> 5;
    int lane = threadIdx.x & 31;
    int lo   = lane & 15;
    int hi   = lane >> 4;

    long long mbase = (long long)blockIdx.x * 128 + wave * 16;
    int rowA = (int)mbase + lo;                 // A-matrix row owned by this lane
    int b = rowA >> 14, pos = rowA & (LL - 1);

    // Center-aligned scatter read: feats[:, w//2 : w//2+Lw, j] = cls_j
    float f0 = (pos >= OFF0 && pos < OFF0 + LW0) ? cls0[(long long)b * LW0 + (pos - OFF0)] : 0.f;
    float f1 = (pos >= OFF1 && pos < OFF1 + LW1) ? cls1[(long long)b * LW1 + (pos - OFF1)] : 0.f;
    float f2 = (pos >= OFF2 && pos < OFF2 + LW2) ? cls2[(long long)b * LW2 + (pos - OFF2)] : 0.f;

    // Layer 1 + GELU directly into A fragments.
    // 16-bit A 16x32 layout: element e -> K = kk*32 + (e<8 ? e : e+8) + hi*8.
    v16h afrag[4];
    #pragma unroll
    for (int kk = 0; kk < 4; ++kk) {
        #pragma unroll
        for (int e = 0; e < 16; ++e) {
            int k = kk * 32 + (e < 8 ? e : e + 8) + hi * 8;
            float h = f0 * sw1[k] + f1 * sw1[HH + k] + f2 * sw1[2 * HH + k] + sb1[k];
            afrag[kk][e] = (_Float16)gelu_exact(h);
        }
    }

    const v16h* bfrags = (const v16h*)w2f;
    #pragma unroll
    for (int nt = 0; nt < 8; ++nt) {
        v8f acc = {0.f, 0.f, 0.f, 0.f, 0.f, 0.f, 0.f, 0.f};
        #pragma unroll
        for (int kk = 0; kk < 4; ++kk) {
            v16h bf = bfrags[(kk * 8 + nt) * 32 + lane];
            acc = __builtin_amdgcn_wmma_f32_16x16x32_f16(
                false, afrag[kk], false, bf, (short)0, acc, false, false);
        }
        int n = nt * 16 + lo;
        float bias = sb2[n];
        // D layout: VGPR r -> row (r + hi*8), col = lo within tile
        #pragma unroll
        for (int r = 0; r < 8; ++r) {
            long long row = mbase + r + hi * 8;
            out[row * HH + n] = acc[r] + bias;
        }
    }
}

// ---------------------------------------------------------------------------
extern "C" void kernel_launch(void* const* d_in, const int* in_sizes, int n_in,
                              void* d_out, int out_size, void* d_ws, size_t ws_size,
                              hipStream_t stream) {
    const float* x     = (const float*)d_in[0];
    const float* cw0   = (const float*)d_in[1];
    const float* cw1   = (const float*)d_in[2];
    const float* cw2   = (const float*)d_in[3];
    const float* convb = (const float*)d_in[4];
    const float* clsw  = (const float*)d_in[5];
    const float* clsb  = (const float*)d_in[6];
    const float* regw  = (const float*)d_in[7];
    const float* regb  = (const float*)d_in[8];
    const float* pw1   = (const float*)d_in[9];
    const float* pb1   = (const float*)d_in[10];
    const float* pw2   = (const float*)d_in[11];
    const float* pb2   = (const float*)d_in[12];
    const float* lw1   = (const float*)d_in[13];
    const float* lb1   = (const float*)d_in[14];
    const float* lw2   = (const float*)d_in[15];
    const float* lb2   = (const float*)d_in[16];

    float* out         = (float*)d_out;
    float* local_feat  = out;                           // [32,16384,128]
    float* global_feat = out + 67108864LL;              // [32,128]
    float* cls0 = global_feat + 4096;                   // [32,16375]
    float* cls1 = cls0 + 32LL * LW0;                    // [32,16371]
    float* cls2 = cls1 + 32LL * LW1;                    // [32,16367]
    float* reg0 = cls2 + 32LL * LW2;
    float* reg1 = reg0 + 32LL * LW0;
    float* reg2 = reg1 + 32LL * LW1;
    float* maskp = reg2 + 32LL * LW2;                   // [32,16384] (all ones)

    float* gfeat = (float*)d_ws;                        // [32,9] pooled features

    conv_heads_kernel<<<2048, 256, 0, stream>>>(x, cw0, cw1, cw2, convb, clsw, clsb,
                                                regw, regb, cls0, cls1, cls2,
                                                reg0, reg1, reg2, maskp);
    pool_kernel<<<96, 256, 0, stream>>>(cls0, cls1, cls2, gfeat);
    gmlp_kernel<<<1, 256, 0, stream>>>(gfeat, pw1, pb1, pw2, pb2, global_feat);
    local_mlp_wmma_kernel<<<4096, 256, 0, stream>>>(cls0, cls1, cls2,
                                                    lw1, lb1, lw2, lb2, local_feat);
}
